// CausalSelfAttention_48713519072140
// MI455X (gfx1250) — matrix operand order repaired and checked by reading, not compiled, not run on hardware
//
#include <hip/hip_runtime.h>

typedef unsigned short u16;
typedef __attribute__((ext_vector_type(16))) __bf16 bf16x16;
typedef __attribute__((ext_vector_type(8)))  float  f32x8;

union FragB16 {
  uint4   u[2];
  bf16x16 v;
};

__device__ __forceinline__ u16 f32_to_bf16(float f) {
  unsigned u = __float_as_uint(f);
  unsigned r = (u + 0x7FFFu + ((u >> 16) & 1u)) >> 16;  // round-nearest-even
  return (u16)r;
}
__device__ __forceinline__ unsigned pack_bf16x2(float a, float b) {
  return (unsigned)f32_to_bf16(a) | ((unsigned)f32_to_bf16(b) << 16);
}
__device__ __forceinline__ f32x8 wmma_bf16(bf16x16 a, bf16x16 b, f32x8 c) {
  return __builtin_amdgcn_wmma_f32_16x16x32_bf16(false, a, false, b, (short)0, c,
                                                 false, false);
}
__device__ __forceinline__ f32x8 zero8() {
  f32x8 z = {0.f, 0.f, 0.f, 0.f, 0.f, 0.f, 0.f, 0.f};
  return z;
}

// Problem constants
#define MROWS 8192   // B*T
#define CMODEL 1024
#define NQKV 3072
#define TSEQ 2048
#define NHEAD 16
#define HDIM 64
#define QKV_ELEMS ((size_t)MROWS * CMODEL)  // 8388608 per tensor

// ---------------------------------------------------------------------------
// Kernel 1: QKV projection.  X[8192,1024] f32  @  W[1024,3072] f32  + bias
// -> Q,K,V bf16 in [B*H, T, 64] layout.
// 256 threads (8 waves), 128x128 tile, BK=32.
// ---------------------------------------------------------------------------
__global__ __launch_bounds__(256) void qkv_gemm_kernel(
    const float* __restrict__ X, const float* __restrict__ W,
    const float* __restrict__ bias, u16* __restrict__ qws,
    u16* __restrict__ kws, u16* __restrict__ vws) {
  __shared__ alignas(16) u16 As[128 * 40];
  __shared__ alignas(16) u16 Bs[128 * 40];

  const int tid = threadIdx.x;
  const int wave = tid >> 5, lane = tid & 31;
  const int lane16 = lane & 15, hf = lane >> 4;
  const int wm = wave >> 2, wn = wave & 3;
  const int bm = blockIdx.x, bn = blockIdx.y;

  f32x8 acc[4][2];
#pragma unroll
  for (int i = 0; i < 4; ++i)
#pragma unroll
    for (int j = 0; j < 2; ++j) acc[i][j] = zero8();

  const float4* Xf4 = reinterpret_cast<const float4*>(X);
  const float4* Wf4 = reinterpret_cast<const float4*>(W);

  for (int kt = 0; kt < CMODEL / 32; ++kt) {
    // Stage A tile 128x32 f32 -> bf16 LDS (row-major, stride 40)
#pragma unroll
    for (int i = 0; i < 4; ++i) {
      int j = tid + i * 256;           // 1024 float4
      int r = j >> 3, c = j & 7;       // 8 float4 per row
      float4 f = Xf4[(size_t)(bm * 128 + r) * 256 + kt * 8 + c];
      uint2 p;
      p.x = pack_bf16x2(f.x, f.y);
      p.y = pack_bf16x2(f.z, f.w);
      *reinterpret_cast<uint2*>(&As[r * 40 + c * 4]) = p;
    }
    // Stage B tile 32x128 f32 -> bf16 LDS transposed: Bs[n][k], stride 40
#pragma unroll
    for (int i = 0; i < 4; ++i) {
      int j = tid + i * 256;           // 1024 float4
      int k = j >> 5, c4 = j & 31;     // 32 float4 per k-row
      float4 f = Wf4[(size_t)(kt * 32 + k) * (NQKV / 4) + bn * 32 + c4];
      int n0 = c4 * 4;
      Bs[(n0 + 0) * 40 + k] = f32_to_bf16(f.x);
      Bs[(n0 + 1) * 40 + k] = f32_to_bf16(f.y);
      Bs[(n0 + 2) * 40 + k] = f32_to_bf16(f.z);
      Bs[(n0 + 3) * 40 + k] = f32_to_bf16(f.w);
    }
    if (kt + 1 < CMODEL / 32) {
      __builtin_prefetch(&Xf4[(size_t)(bm * 128 + (tid >> 3)) * 256 + (kt + 1) * 8], 0, 1);
      __builtin_prefetch(&Wf4[(size_t)((kt + 1) * 32 + (tid >> 5)) * (NQKV / 4) + bn * 32 + (tid & 31)], 0, 1);
    }
    __syncthreads();

    FragB16 a[4], b[2];
#pragma unroll
    for (int mt = 0; mt < 4; ++mt) {
      int row = wm * 64 + mt * 16 + lane16;
      const u16* p = &As[row * 40 + hf * 8];
      a[mt].u[0] = *reinterpret_cast<const uint4*>(p);
      a[mt].u[1] = *reinterpret_cast<const uint4*>(p + 16);
    }
#pragma unroll
    for (int nt = 0; nt < 2; ++nt) {
      int n = wn * 32 + nt * 16 + lane16;
      const u16* p = &Bs[n * 40 + hf * 8];
      b[nt].u[0] = *reinterpret_cast<const uint4*>(p);
      b[nt].u[1] = *reinterpret_cast<const uint4*>(p + 16);
    }
#pragma unroll
    for (int mt = 0; mt < 4; ++mt)
#pragma unroll
      for (int nt = 0; nt < 2; ++nt)
        acc[mt][nt] = wmma_bf16(a[mt].v, b[nt].v, acc[mt][nt]);
    __syncthreads();
  }

  // Epilogue: add bias, route columns to Q/K/V bf16 tensors [B*H, T, 64]
#pragma unroll
  for (int mt = 0; mt < 4; ++mt) {
#pragma unroll
    for (int nt = 0; nt < 2; ++nt) {
      int col = bn * 128 + wn * 32 + nt * 16 + lane16;
      float bb = bias[col];
      int which = col >> 10;
      int c = col & 1023;
      int h = c >> 6, d = c & 63;
      u16* dst = (which == 0) ? qws : ((which == 1) ? kws : vws);
#pragma unroll
      for (int r = 0; r < 8; ++r) {
        int row = bm * 128 + wm * 64 + mt * 16 + r + 8 * hf;
        int bidx = row >> 11, t = row & 2047;
        size_t oidx = ((((size_t)bidx * NHEAD + h) * TSEQ) + t) * HDIM + d;
        dst[oidx] = f32_to_bf16(acc[mt][nt][r] + bb);
      }
    }
  }
}

// ---------------------------------------------------------------------------
// Kernel 2: causal flash attention per (b,h).  4 waves, each owns a 16-row
// query tile; WG covers 64 queries.  KV streamed in chunks of 32.
// Output: bf16 [B*T, 1024] with heads re-merged.
// ---------------------------------------------------------------------------
__global__ __launch_bounds__(128) void attn_kernel(
    const u16* __restrict__ qws, const u16* __restrict__ kws,
    const u16* __restrict__ vws, u16* __restrict__ ows) {
  __shared__ alignas(16) u16 Ks[32 * 72];   // K chunk row-major [kv][d]
  __shared__ alignas(16) u16 Vs[64 * 40];   // V chunk transposed [d][kv]
  __shared__ alignas(16) u16 Pw[4 * 16 * 40];  // per-wave P tile [16][32]

  const int tid = threadIdx.x;
  const int wave = tid >> 5, lane = tid & 31;
  const int lane16 = lane & 15, hf = lane >> 4;
  const int bh = blockIdx.x;          // b*16 + h
  const int qb = blockIdx.y * 64;
  const int q0w = qb + wave * 16;
  const int q_hi = q0w + 15;
  const float SCALE = 0.125f;  // 1/sqrt(64)

  // Preload Q fragments (Dh=64 -> two k-steps of 32)
  FragB16 qf[2];
  {
    const u16* qbase = qws + ((size_t)bh * TSEQ + q0w + lane16) * HDIM;
#pragma unroll
    for (int ks = 0; ks < 2; ++ks) {
      int kb = ks * 32 + hf * 8;
      qf[ks].u[0] = *reinterpret_cast<const uint4*>(qbase + kb);
      qf[ks].u[1] = *reinterpret_cast<const uint4*>(qbase + kb + 16);
    }
  }

  float m_r[8], l_r[8];
  f32x8 Oacc[4];
#pragma unroll
  for (int r = 0; r < 8; ++r) { m_r[r] = -1e30f; l_r[r] = 0.f; }
#pragma unroll
  for (int nt = 0; nt < 4; ++nt) Oacc[nt] = zero8();

  for (int kv0 = 0; kv0 < qb + 64; kv0 += 32) {
    // Stage K chunk: 32x64 bf16, direct copy (uint4 = 8 bf16)
#pragma unroll
    for (int i = 0; i < 2; ++i) {
      int j = tid + i * 128;           // 256 uint4
      int r = j >> 3, c8 = j & 7;
      const u16* src = kws + ((size_t)bh * TSEQ + kv0 + r) * HDIM + c8 * 8;
      *reinterpret_cast<uint4*>(&Ks[r * 72 + c8 * 8]) =
          *reinterpret_cast<const uint4*>(src);
    }
    // Stage V chunk transposed: Vs[d][kv]
#pragma unroll
    for (int i = 0; i < 8; ++i) {
      int j = tid + i * 128;           // 1024 bf16 pairs
      int kvr = j >> 5, p = j & 31;
      int d = p * 2;
      unsigned u = *reinterpret_cast<const unsigned*>(
          vws + ((size_t)bh * TSEQ + kv0 + kvr) * HDIM + d);
      Vs[d * 40 + kvr] = (u16)(u & 0xffffu);
      Vs[(d + 1) * 40 + kvr] = (u16)(u >> 16);
    }
    __syncthreads();

    if (kv0 <= q_hi) {
      // S = Q * K^T  (two 16-wide kv subtiles)
      f32x8 Sacc[2];
#pragma unroll
      for (int s = 0; s < 2; ++s) {
        f32x8 c = zero8();
#pragma unroll
        for (int ks = 0; ks < 2; ++ks) {
          FragB16 bf;
          const u16* bp = &Ks[(s * 16 + lane16) * 72 + ks * 32 + hf * 8];
          bf.u[0] = *reinterpret_cast<const uint4*>(bp);
          bf.u[1] = *reinterpret_cast<const uint4*>(bp + 16);
          c = wmma_bf16(qf[ks].v, bf.v, c);
        }
        Sacc[s] = c;
      }

      // Scale + causal mask + row max
      float sv0[8], sv1[8], tmax[8];
#pragma unroll
      for (int r = 0; r < 8; ++r) {
        int row = q0w + r + 8 * hf;
        float a0 = Sacc[0][r] * SCALE;
        float a1 = Sacc[1][r] * SCALE;
        if (kv0 + lane16 > row) a0 = -1e30f;
        if (kv0 + 16 + lane16 > row) a1 = -1e30f;
        sv0[r] = a0; sv1[r] = a1;
        float tm = fmaxf(a0, a1);
#pragma unroll
        for (int off = 8; off >= 1; off >>= 1)
          tm = fmaxf(tm, __shfl_xor(tm, off, 32));
        tmax[r] = tm;
      }

      // Online softmax update + P to LDS (bf16, A-fragment-friendly layout)
      float fsc[8];
#pragma unroll
      for (int r = 0; r < 8; ++r) {
        float mn = fmaxf(m_r[r], tmax[r]);
        fsc[r] = __expf(m_r[r] - mn);
        m_r[r] = mn;
        float p0 = __expf(sv0[r] - mn);
        float p1 = __expf(sv1[r] - mn);
        float ts = p0 + p1;
#pragma unroll
        for (int off = 8; off >= 1; off >>= 1) ts += __shfl_xor(ts, off, 32);
        l_r[r] = l_r[r] * fsc[r] + ts;
        int prow = r + 8 * hf;
        Pw[wave * 640 + prow * 40 + lane16] = f32_to_bf16(p0);
        Pw[wave * 640 + prow * 40 + 16 + lane16] = f32_to_bf16(p1);
      }
#pragma unroll
      for (int nt = 0; nt < 4; ++nt)
#pragma unroll
        for (int r = 0; r < 8; ++r) Oacc[nt][r] *= fsc[r];

      asm volatile("" ::: "memory");  // keep P stores before P reloads

      // O += P * V
      FragB16 pa;
      const u16* pp = &Pw[wave * 640 + lane16 * 40 + hf * 8];
      pa.u[0] = *reinterpret_cast<const uint4*>(pp);
      pa.u[1] = *reinterpret_cast<const uint4*>(pp + 16);
#pragma unroll
      for (int nt = 0; nt < 4; ++nt) {
        FragB16 vb;
        const u16* vp = &Vs[(nt * 16 + lane16) * 40 + hf * 8];
        vb.u[0] = *reinterpret_cast<const uint4*>(vp);
        vb.u[1] = *reinterpret_cast<const uint4*>(vp + 16);
        Oacc[nt] = wmma_bf16(pa.v, vb.v, Oacc[nt]);
      }
    }
    __syncthreads();
  }

  // Normalize and store: ows[b*T + t][h*64 + d] (bf16)
  const int bidx = bh >> 4, h = bh & 15;
#pragma unroll
  for (int nt = 0; nt < 4; ++nt) {
    int d = nt * 16 + lane16;
#pragma unroll
    for (int r = 0; r < 8; ++r) {
      int row = q0w + r + 8 * hf;
      float o = Oacc[nt][r] / l_r[r];
      ows[((size_t)(bidx * TSEQ + row)) * CMODEL + h * HDIM + d] = f32_to_bf16(o);
    }
  }
}

// ---------------------------------------------------------------------------
// Kernel 3: output projection.  A (bf16) [8192,1024] @ Wout[1024,1024] + bias
// -> f32 d_out.
// ---------------------------------------------------------------------------
__global__ __launch_bounds__(256) void out_gemm_kernel(
    const u16* __restrict__ A, const float* __restrict__ W,
    const float* __restrict__ bias, float* __restrict__ out) {
  __shared__ alignas(16) u16 As[128 * 40];
  __shared__ alignas(16) u16 Bs[128 * 40];

  const int tid = threadIdx.x;
  const int wave = tid >> 5, lane = tid & 31;
  const int lane16 = lane & 15, hf = lane >> 4;
  const int wm = wave >> 2, wn = wave & 3;
  const int bm = blockIdx.x, bn = blockIdx.y;

  f32x8 acc[4][2];
#pragma unroll
  for (int i = 0; i < 4; ++i)
#pragma unroll
    for (int j = 0; j < 2; ++j) acc[i][j] = zero8();

  const float4* Wf4 = reinterpret_cast<const float4*>(W);

  for (int kt = 0; kt < CMODEL / 32; ++kt) {
    // Stage A tile 128x32 bf16 (direct copy)
#pragma unroll
    for (int i = 0; i < 2; ++i) {
      int j = tid + i * 256;           // 512 uint4
      int r = j >> 2, c8 = j & 3;
      const u16* src = A + (size_t)(bm * 128 + r) * CMODEL + kt * 32 + c8 * 8;
      *reinterpret_cast<uint4*>(&As[r * 40 + c8 * 8]) =
          *reinterpret_cast<const uint4*>(src);
    }
    // Stage B tile 32x128 f32 -> bf16 transposed
#pragma unroll
    for (int i = 0; i < 4; ++i) {
      int j = tid + i * 256;
      int k = j >> 5, c4 = j & 31;
      float4 f = Wf4[(size_t)(kt * 32 + k) * (CMODEL / 4) + bn * 32 + c4];
      int n0 = c4 * 4;
      Bs[(n0 + 0) * 40 + k] = f32_to_bf16(f.x);
      Bs[(n0 + 1) * 40 + k] = f32_to_bf16(f.y);
      Bs[(n0 + 2) * 40 + k] = f32_to_bf16(f.z);
      Bs[(n0 + 3) * 40 + k] = f32_to_bf16(f.w);
    }
    __syncthreads();

    FragB16 a[4], b[2];
#pragma unroll
    for (int mt = 0; mt < 4; ++mt) {
      int row = wm * 64 + mt * 16 + lane16;
      const u16* p = &As[row * 40 + hf * 8];
      a[mt].u[0] = *reinterpret_cast<const uint4*>(p);
      a[mt].u[1] = *reinterpret_cast<const uint4*>(p + 16);
    }
#pragma unroll
    for (int nt = 0; nt < 2; ++nt) {
      int n = wn * 32 + nt * 16 + lane16;
      const u16* p = &Bs[n * 40 + hf * 8];
      b[nt].u[0] = *reinterpret_cast<const uint4*>(p);
      b[nt].u[1] = *reinterpret_cast<const uint4*>(p + 16);
    }
#pragma unroll
    for (int mt = 0; mt < 4; ++mt)
#pragma unroll
      for (int nt = 0; nt < 2; ++nt)
        acc[mt][nt] = wmma_bf16(a[mt].v, b[nt].v, acc[mt][nt]);
    __syncthreads();
  }

#pragma unroll
  for (int mt = 0; mt < 4; ++mt) {
#pragma unroll
    for (int nt = 0; nt < 2; ++nt) {
      int col = bn * 128 + wn * 32 + nt * 16 + lane16;
      float bb = bias[col];
#pragma unroll
      for (int r = 0; r < 8; ++r) {
        int row = bm * 128 + wm * 64 + mt * 16 + r + 8 * hf;
        out[(size_t)row * CMODEL + col] = acc[mt][nt][r] + bb;
      }
    }
  }
}

// ---------------------------------------------------------------------------
extern "C" void kernel_launch(void* const* d_in, const int* in_sizes, int n_in,
                              void* d_out, int out_size, void* d_ws,
                              size_t ws_size, hipStream_t stream) {
  const float* x     = (const float*)d_in[0];
  const float* w_qkv = (const float*)d_in[1];
  const float* b_qkv = (const float*)d_in[2];
  const float* w_out = (const float*)d_in[3];
  const float* b_out = (const float*)d_in[4];
  float* out = (float*)d_out;

  u16* qws = (u16*)d_ws;
  u16* kws = qws + QKV_ELEMS;
  u16* vws = kws + QKV_ELEMS;
  u16* aws = vws + QKV_ELEMS;

  qkv_gemm_kernel<<<dim3(MROWS / 128, NQKV / 128), 256, 0, stream>>>(
      x, w_qkv, b_qkv, qws, kws, vws);
  attn_kernel<<<dim3(4 * NHEAD, TSEQ / 64), 128, 0, stream>>>(qws, kws, vws, aws);
  out_gemm_kernel<<<dim3(MROWS / 128, CMODEL / 128), 256, 0, stream>>>(
      aws, w_out, b_out, out);
}
